// MC_LeNet_19653770347000
// MI455X (gfx1250) — matrix operand order repaired and checked by reading, not compiled, not run on hardware
//
#include <hip/hip_runtime.h>
#include <hip/hip_bf16.h>

// ---------------------------------------------------------------------------
// MC-LeNet forward for MI455X (gfx1250, wave32).
// conv1(f32 VALU) -> MC pool -> conv2 (implicit GEMM, v_wmma_f32_16x16x32_f16,
// M=32/wave, block-shared LDS-staged weights, double-buffered) -> MC pool ->
// FC1 (WMMA, M=32/wave) -> FC2+FC3 (f32 VALU).
// Workspace layout (bytes):
//   [0)          w2pack   f16  25*8*16*32 [tap][nt][n][k] = 204,800
//   [204800)     wf1pack  f16  8*64*16*32                 = 524,288
//   [729088)     pooled1  f16  8192*144*32 (chan-last)    = 75,497,472
//   [76226560)   pooled2  f16  8192*2048 (flatten)        = 33,554,432
//   [109780992)  h1       f32  8192*120                   = 3,932,160
// total ~113.7 MB.
// ---------------------------------------------------------------------------

typedef _Float16 v16h __attribute__((ext_vector_type(16)));
typedef _Float16 v8h  __attribute__((ext_vector_type(8)));
typedef float    v8f  __attribute__((ext_vector_type(8)));

__device__ __forceinline__ unsigned pcg_hash(unsigned v) {
  unsigned s = v * 747796405u + 2891336453u;
  unsigned w = ((s >> ((s >> 28u) + 4u)) ^ s) * 277803737u;
  return (w >> 22u) ^ w;
}
__device__ __forceinline__ float urand01(unsigned id, unsigned salt) {
  return (float)(pcg_hash(id ^ salt) >> 8) * (1.0f / 16777216.0f);
}
// categorical over 4 non-negative values (+eps guard, matches reference)
__device__ __forceinline__ int cat4(float v0, float v1, float v2, float v3, float u) {
  const float e = 1e-20f;
  v0 += e; v1 += e; v2 += e; v3 += e;
  float t  = u * (v0 + v1 + v2 + v3);
  float c0 = v0, c1 = c0 + v1, c2 = c1 + v2;
  return (t < c0) ? 0 : (t < c1) ? 1 : (t < c2) ? 2 : 3;
}

__device__ __forceinline__ v16h make_frag(const _Float16* lo, const _Float16* hi) {
  v8h a = *(const v8h*)lo;
  v8h b = *(const v8h*)hi;
  v16h f;
#pragma unroll
  for (int i = 0; i < 8; ++i) { f[i] = a[i]; f[8 + i] = b[i]; }
  return f;
}

// ---------------- weight packing --------------------------------------------
// w2 [128][32][5][5] f32 -> w2pack f16 [tap=25][nt=8][n=16][k=32] (k = ic)
// (tap-major so one tap slice = contiguous 8 KB for block-level LDS staging)
__global__ __launch_bounds__(256) void pack_w2_kernel(const float* __restrict__ w2,
                                                      _Float16* __restrict__ w2p) {
  unsigned tid = blockIdx.x * 256u + threadIdx.x;    // < 102400
  unsigned ic  = tid & 31u;
  unsigned n   = (tid >> 5) & 15u;
  unsigned nt  = (tid >> 9) & 7u;
  unsigned tap = tid >> 12;                          // 0..24
  unsigned oc  = nt * 16u + n;
  w2p[tid] = (_Float16)w2[(((size_t)oc * 32u + ic) * 25u) + tap];
}

// wf1 [2048][120] f32 -> wf1pack f16 [nt=8][chunk=64][n=16][k=32], cols>=120 -> 0
__global__ __launch_bounds__(256) void pack_wf1_kernel(const float* __restrict__ wf1,
                                                       _Float16* __restrict__ wf1p) {
  unsigned tid = blockIdx.x * 256u + threadIdx.x;    // < 262144
  unsigned k     = tid & 31u;
  unsigned n     = (tid >> 5) & 15u;
  unsigned chunk = (tid >> 9) & 63u;
  unsigned nt    = tid >> 15;
  unsigned col   = nt * 16u + n;
  unsigned kg    = chunk * 32u + k;
  float v = (col < 120u) ? wf1[(size_t)kg * 120u + col] : 0.0f;
  wf1p[tid] = (_Float16)v;
}

// ---------------- conv1 + ReLU + MC pool ------------------------------------
// one wave per (batch, pooled pos); lane = output channel 0..31
// writes pooled1 channels-last f16: p1[(b*144+pos)*32 + c]
__global__ __launch_bounds__(256) void conv1_pool_kernel(const float* __restrict__ x,
                                                         const float* __restrict__ w1,
                                                         const float* __restrict__ b1,
                                                         _Float16* __restrict__ p1) {
  unsigned wave = (blockIdx.x * 256u + threadIdx.x) >> 5;  // < 8192*144
  unsigned c    = threadIdx.x & 31u;
  unsigned b    = wave / 144u;
  unsigned pos  = wave % 144u;
  unsigned ph = pos / 12u, pw = pos % 12u;

  float wgt[25];
#pragma unroll
  for (int t = 0; t < 25; ++t) wgt[t] = w1[c * 25u + t];

  const float* xb = x + (size_t)b * 784u + (2u * ph) * 28u + 2u * pw;
  float xin[36];
#pragma unroll
  for (int yy = 0; yy < 6; ++yy)
#pragma unroll
    for (int xx = 0; xx < 6; ++xx) xin[yy * 6 + xx] = xb[yy * 28 + xx];

  float bias = b1[c];
  float v[4];
#pragma unroll
  for (int dy = 0; dy < 2; ++dy)
#pragma unroll
    for (int dx = 0; dx < 2; ++dx) {
      float s = bias;
#pragma unroll
      for (int kh = 0; kh < 5; ++kh)
#pragma unroll
        for (int kw = 0; kw < 5; ++kw)
          s = fmaf(xin[(dy + kh) * 6 + dx + kw], wgt[kh * 5 + kw], s);
      v[dy * 2 + dx] = fmaxf(s, 0.0f);
    }

  unsigned id = (b * 32u + c) * 144u + pos;
  int idx = cat4(v[0], v[1], v[2], v[3], urand01(id, 0x85EBCA6Bu));
  p1[((size_t)b * 144u + pos) * 32u + c] = (_Float16)idx;
}

// ---------------- conv2 (WMMA implicit GEMM) + ReLU + MC pool ---------------
// wave tile: M=32 (batch b, output rows oh = 4*half .. 4*half+3, 8 cols each),
// N=128 (8 ntiles). K = 800 = 25 taps x 32 ic, chunked K=32.
// Weights: one 8 KB tap-slice staged in LDS per block (8 waves share it),
// double-buffered + prefetched -> 8x less L2 weight traffic.
__global__ __launch_bounds__(256) void conv2_wmma_pool_kernel(const _Float16* __restrict__ p1,
                                                              const _Float16* __restrict__ w2p,
                                                              const float* __restrict__ b2,
                                                              _Float16* __restrict__ p2) {
  // B staging: [tap parity][128 rows x 80B], row = (nt*16+n), 64B data + 16B pad
  __shared__ uint4 bstage[2][640];                 // 2 x 10,240 B
  __shared__ float smem[8][16 * 17];               // per-wave pooling tile

  unsigned wv   = threadIdx.x >> 5;
  unsigned lane = threadIdx.x & 31u;
  unsigned wave = blockIdx.x * 8u + wv;            // < 16384
  unsigned b    = wave >> 1;
  unsigned half = wave & 1u;                       // m-tiles: ts = 2*half, 2*half+1
  unsigned nl = lane & 15u, g = lane >> 4;
  unsigned ohl = nl >> 3, ow = nl & 7u;            // A row m = nl -> (ohl, ow)

  // staging assignment: 256 threads x 32B each per 8 KB slice
  unsigned srow  = threadIdx.x >> 1;               // 0..127 (row = nt*16+n)
  unsigned shalf = threadIdx.x & 1u;
  unsigned sidx  = srow * 4u + shalf * 2u;         // uint4 idx within slice (512)
  unsigned didx  = srow * 5u + shalf * 2u;         // uint4 idx within LDS buffer
  const uint4* wsrc = (const uint4*)w2p;

  v8f zero8 = {0.f, 0.f, 0.f, 0.f, 0.f, 0.f, 0.f, 0.f};
  v8f acc0[8], acc1[8];
#pragma unroll
  for (int i = 0; i < 8; ++i) { acc0[i] = zero8; acc1[i] = zero8; }

  const _Float16* pimg = p1 + (size_t)b * 144u * 32u;

  // preload tap 0 slice into LDS buffer 0
  {
    uint4 r0 = wsrc[sidx], r1 = wsrc[sidx + 1u];
    bstage[0][didx] = r0; bstage[0][didx + 1u] = r1;
  }

#pragma unroll 5
  for (int tap = 0; tap < 25; ++tap) {
    int p = tap & 1;
    __syncthreads();                               // bstage[p] ready for all waves

    uint4 r0, r1;
    if (tap + 1 < 25) {                            // prefetch next slice (overlaps WMMAs)
      r0 = wsrc[(unsigned)(tap + 1) * 512u + sidx];
      r1 = wsrc[(unsigned)(tap + 1) * 512u + sidx + 1u];
    }

    int kh = tap / 5, kw = tap % 5;
    int ih0 = (int)(4u * half) + (int)ohl + kh;    // tile 0 rows; tile 1 = +2
    int iw  = (int)ow + kw;
    const _Float16* a0 = pimg + (unsigned)(ih0 * 12 + iw) * 32u;
    v16h af0 = make_frag(a0 + g * 8u,        a0 + 16u + g * 8u);
    v16h af1 = make_frag(a0 + 768u + g * 8u, a0 + 784u + g * 8u);  // +2 rows

    const _Float16* bufp = (const _Float16*)&bstage[p][0];
#pragma unroll
    for (int nt = 0; nt < 8; ++nt) {
      const _Float16* bptr = bufp + (unsigned)nt * 640u + nl * 40u + g * 8u;
      v16h bf = make_frag(bptr, bptr + 16);        // 2x ds_load_b128
      acc0[nt] = __builtin_amdgcn_wmma_f32_16x16x32_f16(
          false, af0, false, bf, (short)0, acc0[nt], false, false);
      acc1[nt] = __builtin_amdgcn_wmma_f32_16x16x32_f16(
          false, af1, false, bf, (short)0, acc1[nt], false, false);
    }

    if (tap + 1 < 25) {                            // fill the other buffer
      uint4* d = &bstage[p ^ 1][0];
      d[didx] = r0; d[didx + 1u] = r1;
    }
  }

  // bias + ReLU + 2x2 MC pooling via LDS (cross-lane gather), per m-tile
#pragma unroll
  for (int mt = 0; mt < 2; ++mt) {
    unsigned ts = 2u * half + (unsigned)mt;
#pragma unroll
    for (int nt = 0; nt < 8; ++nt) {
      float bias = b2[nt * 16 + nl];               // this lane holds column n = nl
#pragma unroll
      for (int r = 0; r < 8; ++r) {
        float av = (mt == 0) ? acc0[nt][r] : acc1[nt][r];
        smem[wv][(r + g * 8u) * 17u + nl] = fmaxf(av + bias, 0.0f);  // [m][n]
      }
      __syncthreads();
#pragma unroll
      for (int t = 0; t < 2; ++t) {
        unsigned o  = lane * 2u + t;               // 0..63 : 4 pool cols x 16 ch
        unsigned pc = o >> 4, ch = o & 15u;
        // block order (row-major in 2x2): m = 2pc, 2pc+1, 8+2pc, 9+2pc
        float v0 = smem[wv][(2u * pc) * 17u + ch];
        float v1 = smem[wv][(2u * pc + 1u) * 17u + ch];
        float v2 = smem[wv][(8u + 2u * pc) * 17u + ch];
        float v3 = smem[wv][(9u + 2u * pc) * 17u + ch];
        unsigned c  = nt * 16u + ch;
        unsigned id = (b * 128u + c) * 16u + ts * 4u + pc;
        int idx = cat4(v0, v1, v2, v3, urand01(id, 0xC2B2AE35u));
        // flatten order of reference reshape: [b][c*16 + py*4 + px], py=ts, px=pc
        p2[(size_t)b * 2048u + c * 16u + ts * 4u + pc] = (_Float16)idx;
      }
      __syncthreads();
    }
  }
}

// ---------------- FC1 (WMMA) + bias + ReLU ----------------------------------
// M=8192 in waves of 32 rows (2 tiles), N=120 padded to 128, K=2048 (64 chunks)
__global__ __launch_bounds__(256) void fc1_wmma_kernel(const _Float16* __restrict__ p2,
                                                       const _Float16* __restrict__ wf1p,
                                                       const float* __restrict__ bf1,
                                                       float* __restrict__ h1) {
  unsigned wave = blockIdx.x * 8u + (threadIdx.x >> 5);  // < 256
  unsigned lane = threadIdx.x & 31u;
  unsigned nl = lane & 15u, g = lane >> 4;
  unsigned m0 = wave * 32u;

  v8f zero8 = {0.f, 0.f, 0.f, 0.f, 0.f, 0.f, 0.f, 0.f};
  v8f acc0[8], acc1[8];
#pragma unroll
  for (int i = 0; i < 8; ++i) { acc0[i] = zero8; acc1[i] = zero8; }

  const _Float16* arow0 = p2 + (size_t)(m0 + nl) * 2048u;
  const _Float16* arow1 = arow0 + 16u * 2048u;

#pragma unroll 2
  for (int chunk = 0; chunk < 64; ++chunk) {
    const _Float16* a0 = arow0 + chunk * 32;
    const _Float16* a1 = arow1 + chunk * 32;
    v16h af0 = make_frag(a0 + g * 8u, a0 + 16u + g * 8u);
    v16h af1 = make_frag(a1 + g * 8u, a1 + 16u + g * 8u);

#pragma unroll
    for (int nt = 0; nt < 8; ++nt) {
      const _Float16* bbase = wf1p + (((unsigned)(nt * 64 + chunk) * 16u + nl) * 32u) + g * 8u;
      v16h bf = make_frag(bbase, bbase + 16);
      acc0[nt] = __builtin_amdgcn_wmma_f32_16x16x32_f16(
          false, af0, false, bf, (short)0, acc0[nt], false, false);
      acc1[nt] = __builtin_amdgcn_wmma_f32_16x16x32_f16(
          false, af1, false, bf, (short)0, acc1[nt], false, false);
    }
  }

#pragma unroll
  for (int nt = 0; nt < 8; ++nt) {
    unsigned col = nt * 16u + nl;              // lane holds column n = nl
    if (col < 120u) {
      float bias = bf1[col];
#pragma unroll
      for (int r = 0; r < 8; ++r) {
        unsigned row0 = m0 + r + g * 8u;       // C layout: M = r + (lane/16)*8
        unsigned row1 = row0 + 16u;
        h1[(size_t)row0 * 120u + col] = fmaxf(acc0[nt][r] + bias, 0.0f);
        h1[(size_t)row1 * 120u + col] = fmaxf(acc1[nt][r] + bias, 0.0f);
      }
    }
  }
}

// ---------------- FC2 + ReLU + FC3 (f32 VALU, tiny) -------------------------
__global__ __launch_bounds__(256) void fc23_kernel(const float* __restrict__ h1,
                                                   const float* __restrict__ wf2,
                                                   const float* __restrict__ bf2,
                                                   const float* __restrict__ wf3,
                                                   const float* __restrict__ bf3,
                                                   float* __restrict__ out) {
  unsigned b = blockIdx.x * 256u + threadIdx.x;  // < 8192
  float h2[84];
#pragma unroll
  for (int j = 0; j < 84; ++j) h2[j] = bf2[j];

  const float* hr = h1 + (size_t)b * 120u;
  for (int i = 0; i < 120; ++i) {
    float hv = hr[i];
    const float* wr = wf2 + i * 84;
#pragma unroll
    for (int j = 0; j < 84; ++j) h2[j] = fmaf(hv, wr[j], h2[j]);
  }

  float o[10];
#pragma unroll
  for (int k = 0; k < 10; ++k) o[k] = bf3[k];
  for (int j = 0; j < 84; ++j) {
    float s = fmaxf(h2[j], 0.0f);
    const float* wr = wf3 + j * 10;
#pragma unroll
    for (int k = 0; k < 10; ++k) o[k] = fmaf(s, wr[k], o[k]);
  }
#pragma unroll
  for (int k = 0; k < 10; ++k) out[(size_t)b * 10u + k] = o[k];
}

// ---------------------------------------------------------------------------
extern "C" void kernel_launch(void* const* d_in, const int* in_sizes, int n_in,
                              void* d_out, int out_size, void* d_ws, size_t ws_size,
                              hipStream_t stream) {
  const float* x   = (const float*)d_in[0];
  const float* w1  = (const float*)d_in[1];
  const float* b1  = (const float*)d_in[2];
  const float* w2  = (const float*)d_in[3];
  const float* b2  = (const float*)d_in[4];
  const float* wf1 = (const float*)d_in[5];
  const float* bf1 = (const float*)d_in[6];
  const float* wf2 = (const float*)d_in[7];
  const float* bf2 = (const float*)d_in[8];
  const float* wf3 = (const float*)d_in[9];
  const float* bf3 = (const float*)d_in[10];
  float* out = (float*)d_out;

  char* ws = (char*)d_ws;
  _Float16* w2pack  = (_Float16*)(ws + 0);
  _Float16* wf1pack = (_Float16*)(ws + 204800);
  _Float16* pooled1 = (_Float16*)(ws + 729088);
  _Float16* pooled2 = (_Float16*)(ws + 76226560ull);
  float*    h1      = (float*)   (ws + 109780992ull);

  pack_w2_kernel <<<400,   256, 0, stream>>>(w2,  w2pack);
  pack_wf1_kernel<<<1024,  256, 0, stream>>>(wf1, wf1pack);
  conv1_pool_kernel<<<147456, 256, 0, stream>>>(x, w1, b1, pooled1);
  conv2_wmma_pool_kernel<<<2048, 256, 0, stream>>>(pooled1, w2pack, b2, pooled2);
  fc1_wmma_kernel<<<32,    256, 0, stream>>>(pooled2, wf1pack, bf1, h1);
  fc23_kernel    <<<32,    256, 0, stream>>>(h1, wf2, bf2, wf3, bf3, out);
}
